// ScaledDotProdAttV2_3693671874898
// MI455X (gfx1250) — compile-verified, matrix operand order checked by reading, not run on hardware
//
#include <hip/hip_runtime.h>

// ---------------------------------------------------------------------------
// MI455X (gfx1250) GQA attention block: projections + flash attention + out-proj
// All matmuls via v_wmma_f32_16x16x32_bf16 (wave32 WMMA, f32 accumulate).
// ---------------------------------------------------------------------------

typedef __attribute__((ext_vector_type(16))) __bf16 v16bf;
typedef __attribute__((ext_vector_type(8)))  float  v8f;

#define WMMA_BF16(a, b, c) \
  __builtin_amdgcn_wmma_f32_16x16x32_bf16(false, (a), false, (b), (short)0, (c), false, false)

#define S_LEN 2048
#define HDIM  128
#define NH    32
#define NKV   8
#define BATCH 2

// ---- fragment load helpers -------------------------------------------------
// Load 8 consecutive K-elements (16B of bf16 / 32B of f32) converting to bf16.
__device__ inline void load8(const float* p, __bf16* d) {
  float4 a = *reinterpret_cast<const float4*>(p);
  float4 b = *reinterpret_cast<const float4*>(p + 4);
  d[0] = (__bf16)a.x; d[1] = (__bf16)a.y; d[2] = (__bf16)a.z; d[3] = (__bf16)a.w;
  d[4] = (__bf16)b.x; d[5] = (__bf16)b.y; d[6] = (__bf16)b.z; d[7] = (__bf16)b.w;
}
__device__ inline void load8(const __bf16* p, __bf16* d) {
  union { float4 f; __bf16 b[8]; } u;
  u.f = *reinterpret_cast<const float4*>(p);
#pragma unroll
  for (int i = 0; i < 8; ++i) d[i] = u.b[i];
}

// 16-bit A/B fragment from a K-contiguous row (16x32 tile, one row per lane).
// ISA layout: lane<16 holds K=0..7 (v0-3) and K=16..23 (v4-7); lane>=16 holds
// K=8..15 and K=24..31.  -> two contiguous 8-element chunks per lane.
template <typename T>
__device__ inline v16bf load_frag_k(const T* rowp, int half) {
  __bf16 t[16];
  load8(rowp + half * 8, t);
  load8(rowp + 16 + half * 8, t + 8);
  v16bf v;
#pragma unroll
  for (int i = 0; i < 16; ++i) v[i] = t[i];
  return v;
}

// ---- 16-lane row max: pure-VALU v_permlane16 (xor nets), LDS-free ----------
__device__ inline float pl16(float x, unsigned lo, unsigned hi) {
#if __has_builtin(__builtin_amdgcn_permlane16)
  int v = __builtin_amdgcn_permlane16(__float_as_int(x), __float_as_int(x),
                                      (int)lo, (int)hi, false, false);
  return __int_as_float(v);
#else
  (void)lo; (void)hi;
  return x;
#endif
}
__device__ inline float rowmax16(float t) {
#if __has_builtin(__builtin_amdgcn_permlane16)
  t = fmaxf(t, pl16(t, 0x67452301u, 0xEFCDAB89u));  // xor 1
  t = fmaxf(t, pl16(t, 0x45670123u, 0xCDEF89ABu));  // xor 2
  t = fmaxf(t, pl16(t, 0x01234567u, 0x89ABCDEFu));  // xor 4
  t = fmaxf(t, pl16(t, 0xFEDCBA98u, 0x76543210u));  // xor 8
#else
#pragma unroll
  for (int d = 1; d < 16; d <<= 1) t = fmaxf(t, __shfl_xor(t, d, 32));
#endif
  return t;
}

// ---------------------------------------------------------------------------
// GEMM: C(M,N) = A(M,K) * Bw(N,K)^T.   A: TA (f32 or bf16), Bw: TB.
// MODE: 0 -> f32 row-major C, 1 -> bf16 row-major C, 2 -> bf16 TRANSPOSED C
//        (C^T (N,M), M-contiguous; frag rows are contiguous -> 16B stores).
// Block: 256 threads = 8 waves (2 M x 4 N), wave tile 32x32, K-step 32.
// ---------------------------------------------------------------------------
template <typename TA, typename TB, int MODE>
__global__ __launch_bounds__(256) void gemm_bt(const TA* __restrict__ A,
                                               const TB* __restrict__ Bw,
                                               float* __restrict__ Cf,
                                               __bf16* __restrict__ Cb,
                                               int M, int N, int K) {
  const int lane = threadIdx.x & 31;
  const int wave = threadIdx.x >> 5;
  const int r    = lane & 15;
  const int half = lane >> 4;
  const int wm   = wave >> 2;                 // 0..1
  const int wn   = wave & 3;                  // 0..3
  const int m0   = blockIdx.y * 64 + wm * 32;
  const int n0   = blockIdx.x * 128 + wn * 32;

  v8f c[2][2] = {};
  const TA* arow0 = A  + (size_t)(m0 + r) * K;
  const TA* arow1 = A  + (size_t)(m0 + 16 + r) * K;
  const TB* brow0 = Bw + (size_t)(n0 + r) * K;
  const TB* brow1 = Bw + (size_t)(n0 + 16 + r) * K;

  for (int k = 0; k < K; k += 32) {
    // stream prefetch two K-steps ahead (speculative: OOB silently dropped)
    __builtin_prefetch(arow0 + k + 64, 0, 1);
    __builtin_prefetch(arow1 + k + 64, 0, 1);
    __builtin_prefetch(brow0 + k + 64, 0, 1);
    __builtin_prefetch(brow1 + k + 64, 0, 1);
    v16bf a0 = load_frag_k(arow0 + k, half);
    v16bf a1 = load_frag_k(arow1 + k, half);
    v16bf b0 = load_frag_k(brow0 + k, half);
    v16bf b1 = load_frag_k(brow1 + k, half);
    c[0][0] = WMMA_BF16(a0, b0, c[0][0]);
    c[0][1] = WMMA_BF16(a0, b1, c[0][1]);
    c[1][0] = WMMA_BF16(a1, b0, c[1][0]);
    c[1][1] = WMMA_BF16(a1, b1, c[1][1]);
  }

#pragma unroll
  for (int i = 0; i < 2; ++i)
#pragma unroll
    for (int jn = 0; jn < 2; ++jn) {
      const int row0 = m0 + i * 16 + half * 8;  // C layout: vgpr j -> M = 8*half + j
      const int col  = n0 + jn * 16 + r;        //           lane&15 -> N
      if (MODE == 2) {
        union { float4 f; __bf16 b[8]; } u;
#pragma unroll
        for (int j = 0; j < 8; ++j) u.b[j] = (__bf16)c[i][jn][j];
        *reinterpret_cast<float4*>(Cb + (size_t)col * M + row0) = u.f;
      } else {
#pragma unroll
        for (int j = 0; j < 8; ++j) {
          size_t idx = (size_t)(row0 + j) * N + col;
          if (MODE == 0) Cf[idx] = c[i][jn][j];
          else           Cb[idx] = (__bf16)c[i][jn][j];
        }
      }
    }
}

// ---------------------------------------------------------------------------
// Flash attention, causal, GQA (H=32, KV=8, HD=128).  One wave = 16 q rows.
// V is pre-transposed: Vt[(g*HD + hd) * (B*S) + b*S + key]  (key-contiguous),
// so V B-fragments use the same contiguous 16B-chunk loads as everything else.
// Row sums come from an extra P x ones WMMA; row max via v_permlane16.
// Block = 8 waves; grid = (S/128, H, B).
// ---------------------------------------------------------------------------
__global__ __launch_bounds__(256) void attn_fwd(const __bf16* __restrict__ Q,
                                                const __bf16* __restrict__ Kt,
                                                const __bf16* __restrict__ Vt,
                                                __bf16* __restrict__ O) {
  __shared__ __bf16 plds[8 * 16 * 32];          // per-wave P staging (C->A relayout)
  const int lane = threadIdx.x & 31;
  const int wave = threadIdx.x >> 5;
  const int r    = lane & 15;
  const int half = lane >> 4;
  const int h    = blockIdx.y;
  const int b    = blockIdx.z;
  const int g    = h >> 2;                      // kv head = h / (H/KV)
  const int q0   = (blockIdx.x * 8 + wave) * 16;
  const int qld  = NH * HDIM;                   // 4096
  const int kld  = NKV * HDIM;                  // 1024
  const float scale = 0.08838834764831845f;     // 1/sqrt(128)

  // Resident Q tile: 16 x 128 as 4 A-fragments.
  const __bf16* qrow = Q + (size_t)(b * S_LEN + q0 + r) * qld + h * HDIM;
  v16bf qa[4];
#pragma unroll
  for (int t = 0; t < 4; ++t) qa[t] = load_frag_k(qrow + t * 32, half);

  v16bf onesf;                                  // all-ones B matrix for row sums
#pragma unroll
  for (int i = 0; i < 16; ++i) onesf[i] = (__bf16)1.0f;

  v8f o[8] = {};                                // 16 x 128 f32 accumulator
  v8f lsum = {};                                // row sums (replicated over cols)
  float m8[8];
#pragma unroll
  for (int j = 0; j < 8; ++j) m8[j] = -3.0e38f;

  __bf16* pbuf = plds + wave * (16 * 32);

  for (int kb = 0; kb < q0 + 16; kb += 32) {
    // ---- scores: S = Q K^T (two 16-key C-frags, 8 wmmas) ----
    const __bf16* krow  = Kt + (size_t)(b * S_LEN + kb + r) * kld + g * HDIM;
    const __bf16* krow2 = krow + (size_t)16 * kld;
    v8f s0 = {}, s1 = {};
#pragma unroll
    for (int t = 0; t < 4; ++t) {
      v16bf k0 = load_frag_k(krow  + t * 32, half);
      v16bf k1 = load_frag_k(krow2 + t * 32, half);
      s0 = WMMA_BF16(qa[t], k0, s0);
      s1 = WMMA_BF16(qa[t], k1, s1);
    }

    // ---- scale + causal mask + online softmax (row = 8*half + j) ----
#pragma unroll
    for (int j = 0; j < 8; ++j) {
      const int qr = q0 + half * 8 + j;
      float v0 = s0[j] * scale;
      float v1 = s1[j] * scale;
      if (kb + r > qr)      v0 = -1.0e30f;
      if (kb + 16 + r > qr) v1 = -1.0e30f;
      const float nm    = fmaxf(m8[j], rowmax16(fmaxf(v0, v1)));
      const float alpha = __expf(m8[j] - nm);
      m8[j] = nm;
      lsum[j] *= alpha;
#pragma unroll
      for (int f = 0; f < 8; ++f) o[f][j] *= alpha;
      // stage P (bf16) row-major 16x32 for C-frag -> A-frag relayout
      pbuf[(half * 8 + j) * 32 + r]      = (__bf16)__expf(v0 - nm);
      pbuf[(half * 8 + j) * 32 + 16 + r] = (__bf16)__expf(v1 - nm);
    }
    asm volatile("s_wait_dscnt 0" ::: "memory");
    v16bf pa = load_frag_k(pbuf + r * 32, half);

    // ---- O += P V, lsum += P x 1  (9 wmmas, all-contiguous B loads) ----
#pragma unroll
    for (int f = 0; f < 8; ++f) {
      const __bf16* vrow =
          Vt + (size_t)(g * HDIM + f * 16 + r) * (BATCH * S_LEN) + b * S_LEN;
      v16bf vb = load_frag_k(vrow + kb, half);
      o[f] = WMMA_BF16(pa, vb, o[f]);
    }
    lsum = WMMA_BF16(pa, onesf, lsum);
  }

  float invl[8];
#pragma unroll
  for (int j = 0; j < 8; ++j) invl[j] = 1.0f / lsum[j];
#pragma unroll
  for (int f = 0; f < 8; ++f)
#pragma unroll
    for (int j = 0; j < 8; ++j) {
      const int qr = q0 + half * 8 + j;
      O[(size_t)(b * S_LEN + qr) * qld + h * HDIM + f * 16 + r] =
          (__bf16)(o[f][j] * invl[j]);
    }
}

// ---------------------------------------------------------------------------
extern "C" void kernel_launch(void* const* d_in, const int* in_sizes, int n_in,
                              void* d_out, int out_size, void* d_ws, size_t ws_size,
                              hipStream_t stream) {
  const float* query = (const float*)d_in[0];
  const float* key   = (const float*)d_in[1];
  const float* value = (const float*)d_in[2];
  const float* Wq    = (const float*)d_in[3];
  const float* Wk    = (const float*)d_in[4];
  const float* Wv    = (const float*)d_in[5];
  const float* Wo    = (const float*)d_in[6];
  // d_in[7] = mask: causal mask is applied analytically in attn_fwd.

  const int Bn = BATCH, S = S_LEN, D = 4096;
  const int M   = Bn * S;        // 4096 rows of activations
  const int Nq  = NH * HDIM;     // 4096
  const int Nkv = NKV * HDIM;    // 1024

  char* w = (char*)d_ws;
  __bf16* Qp = (__bf16*)w; w += (size_t)M * Nq  * 2;   // 33.5 MB
  __bf16* Kp = (__bf16*)w; w += (size_t)M * Nkv * 2;   //  8.4 MB
  __bf16* Vp = (__bf16*)w; w += (size_t)M * Nkv * 2;   //  8.4 MB (transposed)
  __bf16* Ao = (__bf16*)w; w += (size_t)M * Nq  * 2;   // 33.5 MB

  dim3 blk(256);
  gemm_bt<float, float, 1>
      <<<dim3(Nq / 128, M / 64), blk, 0, stream>>>(query, Wq, nullptr, Qp, M, Nq, D);
  gemm_bt<float, float, 1>
      <<<dim3(Nkv / 128, M / 64), blk, 0, stream>>>(key, Wk, nullptr, Kp, M, Nkv, D);
  gemm_bt<float, float, 2>   // V projection stored TRANSPOSED for the PV stage
      <<<dim3(Nkv / 128, M / 64), blk, 0, stream>>>(value, Wv, nullptr, Vp, M, Nkv, D);
  attn_fwd<<<dim3(S / 128, NH, Bn), blk, 0, stream>>>(Qp, Kp, Vp, Ao);
  gemm_bt<__bf16, float, 0>
      <<<dim3(D / 128, M / 64), blk, 0, stream>>>(Ao, Wo, (float*)d_out, nullptr, M, D, Nq);
}
